// DeepSet2d_17179869356
// MI455X (gfx1250) — matrix operand order repaired and checked by reading, not compiled, not run on hardware
//
#include <hip/hip_runtime.h>
#include <hip/hip_bf16.h>

// ---------------------------------------------------------------------------
// DeepSet2d fused WMMA implementation for gfx1250 (MI455X, wave32).
//
// Sizes: B=32, C=3, H=W=224, N=50176, HID=128, EM=64, NCLS=10.
// Per-pixel chain:  x[3] -> relu(x@W1+b1)[128] -> (@W2+b2)[64]
//                   -> relu(e@Wol1_top + locContrib)[128] -> (@Wol2+bol2)[64]
//                   -> softplus -> sum over pixels -> head MLP.
// All GEMMs run on v_wmma_f32_16x16x32_bf16 (f32 accumulate).
//
// Round-3 -> Round-4:
//  * Dropped the volatile weight loads: they lowered to serialized
//    flat_load ... scope:SCOPE_SYS (bypassing WGP$/L2 entirely).
//  * Dropped the per-wave tile loop instead: one wave = one tile
//    (grid 12544 x 256, 8 waves/block, block fully inside one batch image),
//    so there is no loop for LICM to hoist 448 VGPRs of fragments across.
//    sched_barrier(0) between layers bounds scheduler-driven prefetch
//    pressure.
//  * Softplus sum-pool: block-level LDS reduction + 64 atomics per block
//    (800K atomics total, 392-deep per-address chains).
// ---------------------------------------------------------------------------

#define NPIX        50176
#define TILES_PER_B 3136            // NPIX/16
#define TOTAL_TILES 100352          // B * TILES_PER_B
#define WPB         8               // waves per block (256 threads)
#define MAIN_BLOCKS (TOTAL_TILES / WPB)   // 12544; 392 blocks per batch image
#define BLOCKS_PER_B (TILES_PER_B / WPB)  // 392
#define ACT_STRIDE  72              // padded row stride (elems) for LDS staging

typedef __attribute__((ext_vector_type(16))) __bf16       v16bf;
typedef __attribute__((ext_vector_type(8)))  float        v8f;
typedef __attribute__((ext_vector_type(4)))  unsigned int v4u;

union Frag {
    v16bf v;
    __bf16 h[16];
    unsigned short u[16];
    v4u q[2];
};

__device__ __forceinline__ v8f wmma_bf16(v16bf a, v16bf b, v8f c) {
    return __builtin_amdgcn_wmma_f32_16x16x32_bf16(
        /*neg_a=*/false, a, /*neg_b=*/false, b,
        /*c_mod=*/(short)0, c, /*reuse_a=*/false, /*reuse_b=*/false);
}

// B-fragment layout in memory: [frag][lane][16 bf16] -> 32 contiguous B/lane.
// Whole weight image is 57KB -> resident in the 64KB WGP$ after first touch.
__device__ __forceinline__ v16bf loadW(const unsigned short* __restrict__ fw,
                                       int f, int lane) {
    const v4u* p = (const v4u*)(fw + ((size_t)(f * 32 + lane) << 4));
    Frag r;
    r.q[0] = p[0];
    r.q[1] = p[1];
    return r.v;
}

// A-fragment (16x32 bf16) from LDS staging buffer (rows = pixels, cols = K).
// Lane L<16 : row L,    slots 0..15 -> K=ks*32+0..15
// Lane L>=16: row L-16, slots 0..15 -> K=ks*32+16..31
__device__ __forceinline__ v16bf loadA(const __bf16* base, int ks, int lane) {
    int row = lane & 15;
    int off = ks * 32 + ((lane >> 4) ? 16 : 0);
    const __bf16* r = base + row * ACT_STRIDE + off;
    Frag fr;
    fr.q[0] = *(const v4u*)(r);
    fr.q[1] = *(const v4u*)(r + 8);
    return fr.v;
}

__device__ __forceinline__ float softplusf(float x) {
    return (x > 15.f) ? x : __logf(1.f + __expf(x));
}

// ---------------------------------------------------------------------------
// Prep kernel 1: repack weights into bf16 B-fragment layout.
// Fragment order: [0..7]   obs W1 (Kpad=32, N=128)
//                 [8..23]  obs W2 (K=128, N=64):  8 + ks*4 + nt
//                 [24..39] ol  W1 rows 0..63 (K=64, N=128): 24 + ks*8 + nt
//                 [40..55] ol  W2 (K=128, N=64): 40 + ks*4 + nt
// In-fragment: lane<16 -> (k=ks*32+s,    n=nt*16+lane)
//              lane>=16-> (k=ks*32+16+s, n=nt*16+lane-16)
// ---------------------------------------------------------------------------
__global__ void prep_weights_kernel(const float* __restrict__ w_obs1,
                                    const float* __restrict__ w_obs2,
                                    const float* __restrict__ w_ol1,
                                    const float* __restrict__ w_ol2,
                                    __bf16* __restrict__ fragW) {
    int tid  = blockIdx.x * blockDim.x + threadIdx.x;   // 0..28671
    int f    = tid >> 9;
    int lane = (tid >> 4) & 31;
    int s    = tid & 15;
    int kl   = (lane < 16) ? s : 16 + s;
    int nl   = lane & 15;

    const float* W; int K, Nn, ks, nt;
    if (f < 8)       { W = w_obs1; K = 3;   Nn = 128; ks = 0;      nt = f;          }
    else if (f < 24) { int g = f - 8;  W = w_obs2; K = 128; Nn = 64;  ks = g >> 2; nt = g & 3; }
    else if (f < 40) { int g = f - 24; W = w_ol1;  K = 64;  Nn = 128; ks = g >> 3; nt = g & 7; }
    else             { int g = f - 40; W = w_ol2;  K = 128; Nn = 64;  ks = g >> 2; nt = g & 3; }

    int k = ks * 32 + kl;
    int n = nt * 16 + nl;
    float v = (k < K) ? W[k * Nn + n] : 0.f;
    fragW[tid] = (__bf16)v;
}

// ---------------------------------------------------------------------------
// Prep kernel 2: per-pixel location contribution, stored in C-fragment layout:
//   loc_ws[((pixTile*8 + ntile)*32 + lane)*8 + k]  (f32)
// where lane = (col&15) + 16*(m>>3), k = m&7, m = pixel within tile.
// contrib = (relu(loc@Wl1+bl1) @ Wl2 + bl2) @ Wol1[64:,:] + b_ol1
// ---------------------------------------------------------------------------
__global__ __launch_bounds__(128) void prep_loc_kernel(
        const float* __restrict__ w_loc1, const float* __restrict__ b_loc1,
        const float* __restrict__ w_loc2, const float* __restrict__ b_loc2,
        const float* __restrict__ w_ol1,  const float* __restrict__ b_ol1,
        float* __restrict__ loc_ws) {
    __shared__ float h[128];
    __shared__ float es[64];
    int n = blockIdx.x;                 // pixel index 0..50175
    int t = threadIdx.x;                // 0..127
    int iy = n / 224, ix = n - iy * 224;
    float fy = -10.f + 20.f * (float)iy / 223.f;
    float fx = -10.f + 20.f * (float)ix / 223.f;

    float a = fy * w_loc1[t] + fx * w_loc1[128 + t] + b_loc1[t];
    h[t] = fmaxf(a, 0.f);
    __syncthreads();
    if (t < 64) {
        float e = b_loc2[t];
        #pragma unroll 8
        for (int i = 0; i < 128; ++i) e += h[i] * w_loc2[i * 64 + t];
        es[t] = e;
    }
    __syncthreads();
    float c = b_ol1[t];
    #pragma unroll 8
    for (int j = 0; j < 64; ++j) c += es[j] * w_ol1[(64 + j) * 128 + t];

    int pt = n >> 4, m = n & 15;
    int nt = t >> 4, cc = t & 15;
    int lane = cc + ((m >> 3) << 4);
    int k = m & 7;
    loc_ws[((size_t)(pt * 8 + nt) * 32 + lane) * 8 + k] = c;
}

__global__ void zero_kernel(float* __restrict__ p, int n) {
    int i = blockIdx.x * blockDim.x + threadIdx.x;
    if (i < n) p[i] = 0.f;
}

// ---------------------------------------------------------------------------
// Main fused kernel: one wave per 16-pixel tile. 56 WMMAs per wave.
// ---------------------------------------------------------------------------
__global__ __launch_bounds__(256) void deepset_main_kernel(
        const float* __restrict__ images,
        const float* __restrict__ b_obs1, const float* __restrict__ b_obs2,
        const float* __restrict__ b_ol2,
        const unsigned short* __restrict__ fragW,
        const float* __restrict__ loc_ws,
        float* __restrict__ em_set) {
    __shared__ __bf16 act[WPB][16 * ACT_STRIDE];
    __shared__ float  red[WPB][32][4];

    const int lane  = threadIdx.x & 31;
    const int wid   = threadIdx.x >> 5;
    __bf16* A = act[wid];

    const int b   = blockIdx.x / BLOCKS_PER_B;                    // batch image
    const int ptn = (blockIdx.x - b * BLOCKS_PER_B) * WPB + wid;  // pixel tile

    const int col0 = lane & 15;     // column within an N-tile (C/D layout)
    const int half = lane >> 4;     // row half: rows half*8 .. half*8+7
    __bf16* Arow = A + half * 8 * ACT_STRIDE + col0;   // store base for C tiles

    // ---- Layer 1 A-fragment: 16 pixels x K=32 (only K=0..2 nonzero) ----
    Frag ax;
    #pragma unroll
    for (int i = 0; i < 16; ++i) ax.u[i] = 0;
    int pix = ptn * 16 + col0;
    if (half == 0) {
        const float* ip = images + (size_t)b * 3 * NPIX + pix;
        ax.h[0] = (__bf16)ip[0];
        ax.h[1] = (__bf16)ip[NPIX];
        ax.h[2] = (__bf16)ip[2 * NPIX];
    }

    // ---- Layer 1: H = relu(X @ W1 + b1), 16x128 (fused store) ----
    #pragma unroll
    for (int nt = 0; nt < 8; ++nt) {
        v8f c = {};
        v8f h = wmma_bf16(ax.v, loadW(fragW, nt, lane), c);
        float bb = b_obs1[nt * 16 + col0];
        #pragma unroll
        for (int k = 0; k < 8; ++k)
            Arow[k * ACT_STRIDE + nt * 16] = (__bf16)fmaxf(h[k] + bb, 0.f);
    }
    __builtin_amdgcn_sched_barrier(0);

    // ---- Layer 2: E = H @ W2 + b2, 16x64 ----
    v8f e[4];
    #pragma unroll
    for (int nt = 0; nt < 4; ++nt) {
        float bb = b_obs2[nt * 16 + col0];
        #pragma unroll
        for (int k = 0; k < 8; ++k) e[nt][k] = bb;
    }
    #pragma unroll
    for (int ks = 0; ks < 4; ++ks) {
        v16bf a = loadA(A, ks, lane);
        #pragma unroll
        for (int nt = 0; nt < 4; ++nt)
            e[nt] = wmma_bf16(a, loadW(fragW, 8 + ks * 4 + nt, lane), e[nt]);
    }
    __builtin_amdgcn_sched_barrier(0);
    #pragma unroll
    for (int nt = 0; nt < 4; ++nt) {
        #pragma unroll
        for (int k = 0; k < 8; ++k)
            Arow[k * ACT_STRIDE + nt * 16] = (__bf16)e[nt][k];
    }
    __builtin_amdgcn_sched_barrier(0);

    // ---- OL layer 1: G = relu(E @ Wol1_top + locContrib), 16x128 ----
    v8f g[8];
    #pragma unroll
    for (int nt = 0; nt < 8; ++nt) {
        const float4* lp = (const float4*)(loc_ws +
            ((size_t)(ptn * 8 + nt) * 32 + lane) * 8);
        float4 l0 = lp[0], l1 = lp[1];
        g[nt][0] = l0.x; g[nt][1] = l0.y; g[nt][2] = l0.z; g[nt][3] = l0.w;
        g[nt][4] = l1.x; g[nt][5] = l1.y; g[nt][6] = l1.z; g[nt][7] = l1.w;
    }
    #pragma unroll
    for (int ks = 0; ks < 2; ++ks) {
        v16bf a = loadA(A, ks, lane);
        #pragma unroll
        for (int nt = 0; nt < 8; ++nt)
            g[nt] = wmma_bf16(a, loadW(fragW, 24 + ks * 8 + nt, lane), g[nt]);
    }
    __builtin_amdgcn_sched_barrier(0);
    #pragma unroll
    for (int nt = 0; nt < 8; ++nt) {
        #pragma unroll
        for (int k = 0; k < 8; ++k)
            Arow[k * ACT_STRIDE + nt * 16] = (__bf16)fmaxf(g[nt][k], 0.f);
    }
    __builtin_amdgcn_sched_barrier(0);

    // ---- OL layer 2: O = G @ Wol2 + bol2; softplus; per-lane partials ----
    v8f o[4];
    #pragma unroll
    for (int nt = 0; nt < 4; ++nt) {
        float bb = b_ol2[nt * 16 + col0];
        #pragma unroll
        for (int k = 0; k < 8; ++k) o[nt][k] = bb;
    }
    #pragma unroll
    for (int ks = 0; ks < 4; ++ks) {
        v16bf a = loadA(A, ks, lane);
        #pragma unroll
        for (int nt = 0; nt < 4; ++nt)
            o[nt] = wmma_bf16(a, loadW(fragW, 40 + ks * 4 + nt, lane), o[nt]);
    }
    #pragma unroll
    for (int nt = 0; nt < 4; ++nt) {
        float s = 0.f;
        #pragma unroll
        for (int k = 0; k < 8; ++k) s += softplusf(o[nt][k]);
        red[wid][lane][nt] = s;
    }

    // ---- Block reduction over 8 waves x 2 lane-halves, then 64 atomics ----
    __syncthreads();
    if (threadIdx.x < 64) {
        int nt = threadIdx.x >> 4, c = threadIdx.x & 15;
        float s = 0.f;
        #pragma unroll
        for (int w = 0; w < WPB; ++w)
            s += red[w][c][nt] + red[w][c + 16][nt];
        atomicAdd(em_set + b * 64 + threadIdx.x, s);
    }
}

// ---------------------------------------------------------------------------
// Classifier head: logits = relu(em_set @ Wc1 + bc1) @ Wc2 + bc2
// ---------------------------------------------------------------------------
__global__ __launch_bounds__(128) void cls_kernel(
        const float* __restrict__ em_set,
        const float* __restrict__ w1, const float* __restrict__ b1,
        const float* __restrict__ w2, const float* __restrict__ b2,
        float* __restrict__ out) {
    __shared__ float h[128];
    __shared__ float es[64];
    int b = blockIdx.x, t = threadIdx.x;
    if (t < 64) es[t] = em_set[b * 64 + t];
    __syncthreads();
    float a = b1[t];
    #pragma unroll 8
    for (int i = 0; i < 64; ++i) a += es[i] * w1[i * 128 + t];
    h[t] = fmaxf(a, 0.f);
    __syncthreads();
    if (t < 10) {
        float a2 = b2[t];
        #pragma unroll 8
        for (int j = 0; j < 128; ++j) a2 += h[j] * w2[j * 10 + t];
        out[b * 10 + t] = a2;
    }
}

// ---------------------------------------------------------------------------
extern "C" void kernel_launch(void* const* d_in, const int* in_sizes, int n_in,
                              void* d_out, int out_size, void* d_ws, size_t ws_size,
                              hipStream_t stream) {
    (void)in_sizes; (void)n_in; (void)out_size; (void)ws_size;
    const float* images = (const float*)d_in[0];
    const float* w_obs1 = (const float*)d_in[1];
    const float* b_obs1 = (const float*)d_in[2];
    const float* w_obs2 = (const float*)d_in[3];
    const float* b_obs2 = (const float*)d_in[4];
    const float* w_loc1 = (const float*)d_in[5];
    const float* b_loc1 = (const float*)d_in[6];
    const float* w_loc2 = (const float*)d_in[7];
    const float* b_loc2 = (const float*)d_in[8];
    const float* w_ol1  = (const float*)d_in[9];
    const float* b_ol1  = (const float*)d_in[10];
    const float* w_ol2  = (const float*)d_in[11];
    const float* b_ol2  = (const float*)d_in[12];
    const float* w_cls1 = (const float*)d_in[13];
    const float* b_cls1 = (const float*)d_in[14];
    const float* w_cls2 = (const float*)d_in[15];
    const float* b_cls2 = (const float*)d_in[16];
    float* out = (float*)d_out;

    // Workspace carve-up (all offsets 256B aligned).
    __bf16* fragW  = (__bf16*)d_ws;                                       // 57,344 B
    float*  loc_ws = (float*)((char*)d_ws + 65536);                       // 25,690,112 B
    float*  em_set = (float*)((char*)d_ws + 65536 + 25690112);            // 8,192 B

    prep_weights_kernel<<<112, 256, 0, stream>>>(w_obs1, w_obs2, w_ol1, w_ol2, fragW);
    zero_kernel<<<8, 256, 0, stream>>>(em_set, 32 * 64);
    prep_loc_kernel<<<NPIX, 128, 0, stream>>>(w_loc1, b_loc1, w_loc2, b_loc2,
                                              w_ol1, b_ol1, loc_ws);
    deepset_main_kernel<<<MAIN_BLOCKS, 256, 0, stream>>>(
        images, b_obs1, b_obs2, b_ol2, (const unsigned short*)fragW, loc_ws, em_set);
    cls_kernel<<<32, 128, 0, stream>>>(em_set, w_cls1, b_cls1, w_cls2, b_cls2, out);
}
